// RNN_84146999263393
// MI455X (gfx1250) — matrix operand order repaired and checked by reading, not compiled, not run on hardware
//
#include <hip/hip_runtime.h>
#include <hip/hip_bf16.h>
#include <math.h>

// ---------------- problem constants ----------------
constexpr int cB = 64, cT = 512, cE = 512, cH = 1024, cO = 512;
constexpr int cBT = cB * cT;          // 32768 rows of the layer-0 projection
constexpr int cHS = cB * cH;          // one hidden-state buffer (elements)
constexpr int NBLK = 32;              // persistent recurrent grid size
constexpr int H_PER_BLK = cH / NBLK;  // 32 hidden columns per workgroup
constexpr int O_PER_BLK = cO / NBLK;  // 16 output columns per workgroup

// LDS weight staging: rows padded by 8 bf16 so row stride = 516 dwords.
// 516 mod 64 = 4 -> lane n (weight row n) starts at bank 4n; a ds_load_b128
// from lanes 0..15 covers all 64 banks conflict-free.
constexpr int LDSW = cH + 8;                       // padded row stride (bf16)
constexpr int LDS_WHH0 = 0;                        // 32 rows
constexpr int LDS_WIH1 = LDS_WHH0 + 32 * LDSW;     // 32 rows
constexpr int LDS_WHH1 = LDS_WIH1 + 32 * LDSW;     // 32 rows
constexpr int LDS_WOUT = LDS_WHH1 + 32 * LDSW;     // 16 rows
constexpr int LDS_ELEMS = LDS_WOUT + 16 * LDSW;    // 115,584 bf16
constexpr size_t LDS_BYTES = (size_t)LDS_ELEMS * 2;  // ~226 KB (<= 320 KB/WGP)

typedef __bf16 bf16_t;
typedef __attribute__((ext_vector_type(16))) __bf16 v16bf;
typedef __attribute__((ext_vector_type(8)))  float  v8f;

#if __has_builtin(__builtin_amdgcn_tanhf)
#define TANHF(x) __builtin_amdgcn_tanhf(x)
#else
#define TANHF(x) tanhf(x)
#endif

// ---------------- WMMA helpers (gfx1250, wave32) ----------------
// D = A(16x32 bf16) * B(32x16 bf16) + C(16x16 f32)
__device__ __forceinline__ v8f wmma_bf16(v16bf a, v16bf b, v8f c) {
  return __builtin_amdgcn_wmma_f32_16x16x32_bf16(
      /*neg_a=*/false, a, /*neg_b=*/false, b,
      /*c_mod=*/(short)0, c, /*reuse_a=*/false, /*reuse_b=*/false);
}

// A fragment, 16(M)x32(K) bf16 from a row-major [rows x ld] matrix.
// Contiguous k order (B matches): lane&15 = M row; lanes 0-15 carry K 0..15,
// lanes 16-31 carry K 16..31. 32 contiguous bytes per lane -> 2x b128.
__device__ __forceinline__ v16bf load_a_bf(const bf16_t* __restrict__ p, int ld,
                                           int row0, int k0) {
  const int lane = threadIdx.x & 31;
  const bf16_t* base =
      p + (size_t)(row0 + (lane & 15)) * ld + (k0 + ((lane >> 4) << 4));
  v16bf f;
#pragma unroll
  for (int i = 0; i < 8; ++i) f[i] = base[i];
#pragma unroll
  for (int i = 0; i < 8; ++i) f[i + 8] = base[i + 8];
  return f;
}

__device__ __forceinline__ v16bf load_a_f32(const float* __restrict__ p, int ld,
                                            int row0, int k0) {
  const int lane = threadIdx.x & 31;
  const float* base =
      p + (size_t)(row0 + (lane & 15)) * ld + (k0 + ((lane >> 4) << 4));
  v16bf f;
#pragma unroll
  for (int i = 0; i < 8; ++i) f[i] = (__bf16)base[i];
#pragma unroll
  for (int i = 0; i < 8; ++i) f[i + 8] = (__bf16)base[i + 8];
  return f;
}

// B fragment, 32(K)x16(N) bf16 (B[k][n] = W[n][k]; column n of B = row n of
// the row-major weight slice). Same k order as A: lane&15 = N col; lanes 0-15 carry
// K 0..15, lanes 16-31 K 16..31 -> 32 contiguous bytes per lane.
__device__ __forceinline__ v16bf load_b_f32(const float* __restrict__ w, int ld,
                                            int nrow0, int k0) {
  const int lane = threadIdx.x & 31;
  const float* base =
      w + (size_t)(nrow0 + (lane & 15)) * ld + (k0 + ((lane >> 4) << 4));
  v16bf f;
#pragma unroll
  for (int i = 0; i < 16; ++i) f[i] = (__bf16)base[i];
  return f;
}

// Same B fragment but sourced from the padded LDS weight cache (ds_load_b128).
__device__ __forceinline__ v16bf load_b_lds(const bf16_t* w, int nrow0, int k0) {
  const int lane = threadIdx.x & 31;
  const bf16_t* base =
      w + (nrow0 + (lane & 15)) * LDSW + (k0 + ((lane >> 4) << 4));
  v16bf f;
#pragma unroll
  for (int i = 0; i < 16; ++i) f[i] = base[i];
  return f;
}

// ---------------- utility kernels ----------------
__global__ void cvt_f32_to_bf16(const float* __restrict__ src,
                                bf16_t* __restrict__ dst, int n) {
  int i = blockIdx.x * blockDim.x + threadIdx.x;
  if (i < n) dst[i] = (bf16_t)src[i];
}

// ---------------- phase 1: xp = emb @ W_ih0^T + b_ih0  (bf16 out) ----------
// One 16x16 tile per wave, 8 waves/block; consecutive waves share the A tile.
__global__ void __launch_bounds__(256)
xp_gemm(const float* __restrict__ emb, const float* __restrict__ Wih0,
        const float* __restrict__ bih0, bf16_t* __restrict__ xp) {
  const int wave = threadIdx.x >> 5;
  const int lane = threadIdx.x & 31;
  const int tile = blockIdx.x * 8 + wave;
  const int nt = tile & (cH / 16 - 1);  // 0..63
  const int mt = tile >> 6;             // 0..2047
  const int m0 = mt * 16, n0 = nt * 16;

  v8f acc = {};
#pragma unroll 4
  for (int k = 0; k < cE; k += 32) {
    v16bf a = load_a_f32(emb, cE, m0, k);
    v16bf b = load_b_f32(Wih0, cE, n0, k);
    acc = wmma_bf16(a, b, acc);
  }
  // C layout: lane&15 = n, row = r + 8*(lane>=16)
  const int n = n0 + (lane & 15);
  const int mofs = (lane >> 4) << 3;
  const float bias = bih0[n];
#pragma unroll
  for (int r = 0; r < 8; ++r) {
    int m = m0 + r + mofs;                        // m = b*T + t
    xp[(size_t)m * cH + n] = (bf16_t)(acc[r] + bias);
  }
}

// ---------------- device-wide barrier (persistent grid of NBLK blocks) -----
__device__ __forceinline__ void grid_barrier(unsigned int* bar, unsigned int gen) {
  __syncthreads();
  if (threadIdx.x == 0) {
    __threadfence();                      // make our h writes visible (L2)
    atomicAdd(bar, 1u);
    const unsigned int target = gen * (unsigned int)NBLK;
    while (atomicAdd(bar, 0u) < target) __builtin_amdgcn_s_sleep(1);
    __threadfence();                      // acquire others' writes
  }
  __syncthreads();
}

// ---------------- phase 2: persistent recurrent kernel ---------------------
// 32 blocks x 256 threads (8 waves), one block per WGP. Block owns 32
// H-columns / 16 O-columns. Its weight slice (226 KB incl. padding) is staged
// into LDS ONCE and reused for all 512 timesteps: B fragments come from LDS
// (conflict-free ds_load_b128), only the small hidden state streams from L2.
// Hidden states double-buffered in workspace; 2 grid barriers per timestep.
__global__ void __launch_bounds__(256, 1)
rnn_recurrent(const bf16_t* __restrict__ xp,
              const bf16_t* __restrict__ Whh0, const bf16_t* __restrict__ Wih1,
              const bf16_t* __restrict__ Whh1, const bf16_t* __restrict__ Wout,
              const float* __restrict__ bhh0, const float* __restrict__ bih1,
              const float* __restrict__ bhh1, const float* __restrict__ bout,
              bf16_t* __restrict__ h0buf, bf16_t* __restrict__ h1buf,
              float* __restrict__ out, unsigned int* __restrict__ bar) {
  extern __shared__ bf16_t lds[];

  const int wave = threadIdx.x >> 5;
  const int lane = threadIdx.x & 31;
  const int n0 = blockIdx.x * H_PER_BLK;   // this block's H slice base
  const int o0 = blockIdx.x * O_PER_BLK;   // this block's O slice base
  const int mt = wave >> 1;                // 0..3 -> M rows (B=64)
  const int nt = wave & 1;                 // 0..1 -> 16-col halves of slice
  const int m0 = mt * 16;
  const int nl0 = nt * 16;                 // N base, local to the LDS slice
  const int nAbs = n0 + nl0 + (lane & 15); // absolute H column for epilogue
  const int mofs = (lane >> 4) << 3;       // +0 / +8 C-row offset

  // ---- stage this block's weight slice into LDS (once) ----
  {
    constexpr int VEC = 8;                 // bf16 per 16-byte vector
    constexpr int VROW = cH / VEC;         // 128 vectors per weight row
    auto stage = [&](const bf16_t* w, int r0, int nrows, int ldsBase) {
      for (int idx = threadIdx.x; idx < nrows * VROW; idx += 256) {
        const int r = idx / VROW, c = (idx % VROW) * VEC;
        *(uint4*)(lds + ldsBase + r * LDSW + c) =
            *(const uint4*)(w + (size_t)(r0 + r) * cH + c);
      }
    };
    stage(Whh0, n0, 32, LDS_WHH0);
    stage(Wih1, n0, 32, LDS_WIH1);
    stage(Whh1, n0, 32, LDS_WHH1);
    stage(Wout, o0, 16, LDS_WOUT);
  }
  __syncthreads();

  unsigned int gen = 0;

  for (int t = 0; t < cT; ++t) {
    const int p = t & 1;
    const bf16_t* h0r = h0buf + (size_t)p * cHS;
    bf16_t*       h0w = h0buf + (size_t)(p ^ 1) * cHS;
    const bf16_t* h1r = h1buf + (size_t)p * cHS;
    bf16_t*       h1w = h1buf + (size_t)(p ^ 1) * cHS;

    // ---- layer 0: h0n = tanh(xp_t + h0 @ Whh0^T + bhh0) ----
    {
      v8f acc = {};
#pragma unroll 4
      for (int k = 0; k < cH; k += 32) {
        v16bf a = load_a_bf(h0r, cH, m0, k);
        v16bf b = load_b_lds(lds + LDS_WHH0, nl0, k);
        acc = wmma_bf16(a, b, acc);
      }
      const float bias = bhh0[nAbs];
#pragma unroll
      for (int r = 0; r < 8; ++r) {
        const int m = m0 + r + mofs;       // batch row
        const float x = (float)xp[((size_t)m * cT + t) * cH + nAbs];
        h0w[(size_t)m * cH + nAbs] = (bf16_t)TANHF(acc[r] + bias + x);
      }
    }
    grid_barrier(bar, ++gen);              // h0n complete everywhere

    // ---- layer 1: h1n = tanh(h0n @ Wih1^T + bih1 + h1 @ Whh1^T + bhh1) ----
    {
      v8f acc = {};
#pragma unroll 4
      for (int k = 0; k < cH; k += 32) {
        v16bf a = load_a_bf(h0w, cH, m0, k);
        v16bf b = load_b_lds(lds + LDS_WIH1, nl0, k);
        acc = wmma_bf16(a, b, acc);
      }
#pragma unroll 4
      for (int k = 0; k < cH; k += 32) {
        v16bf a = load_a_bf(h1r, cH, m0, k);
        v16bf b = load_b_lds(lds + LDS_WHH1, nl0, k);
        acc = wmma_bf16(a, b, acc);
      }
      const float bias = bih1[nAbs] + bhh1[nAbs];
#pragma unroll
      for (int r = 0; r < 8; ++r) {
        const int m = m0 + r + mofs;
        h1w[(size_t)m * cH + nAbs] = (bf16_t)TANHF(acc[r] + bias);
      }
    }
    grid_barrier(bar, ++gen);              // h1n complete everywhere

    // ---- output: out_t = h1n @ Wout^T + bout  (waves 0..3, 4 M-tiles) ----
    if (wave < 4) {
      const int om0 = wave * 16;
      const int oAbs = o0 + (lane & 15);
      v8f acc = {};
#pragma unroll 4
      for (int k = 0; k < cH; k += 32) {
        v16bf a = load_a_bf(h1w, cH, om0, k);
        v16bf b = load_b_lds(lds + LDS_WOUT, 0, k);
        acc = wmma_bf16(a, b, acc);
      }
      const float bias = bout[oAbs];
#pragma unroll
      for (int r = 0; r < 8; ++r) {
        const int m = om0 + r + mofs;
        out[((size_t)m * cT + t) * cO + oAbs] = acc[r] + bias;
      }
    }
    // No extra barrier needed: next step's writes touch the opposite parity
    // buffers, whose last readers all synchronized at the barriers above.
  }
}

// ---------------- host-side launch ----------------
extern "C" void kernel_launch(void* const* d_in, const int* in_sizes, int n_in,
                              void* d_out, int out_size, void* d_ws,
                              size_t ws_size, hipStream_t stream) {
  (void)in_sizes; (void)n_in; (void)out_size; (void)ws_size;

  const float* emb  = (const float*)d_in[0];   // (B,T,E)
  const float* h0in = (const float*)d_in[1];   // (L,B,H)
  const float* Wih0 = (const float*)d_in[2];   // (H,E)
  const float* bih0 = (const float*)d_in[3];
  const float* Wih1 = (const float*)d_in[4];   // (H,H)
  const float* bih1 = (const float*)d_in[5];
  const float* Whh0 = (const float*)d_in[6];   // (H,H)
  const float* bhh0 = (const float*)d_in[7];
  const float* Whh1 = (const float*)d_in[8];   // (H,H)
  const float* bhh1 = (const float*)d_in[9];
  const float* Wout = (const float*)d_in[10];  // (O,H)
  const float* bout = (const float*)d_in[11];
  float* out = (float*)d_out;                  // (B,T,O) f32

  // Workspace carve-up (~75 MB): xp(bf16) + bf16 weights + hidden buffers.
  char* ws = (char*)d_ws;
  size_t off = 0;
  auto take = [&](size_t bytes) -> char* {
    off = (off + 255) & ~(size_t)255;
    char* p = ws + off;
    off += bytes;
    return p;
  };
  bf16_t* xp     = (bf16_t*)take((size_t)cBT * cH * sizeof(bf16_t));
  bf16_t* whh0b  = (bf16_t*)take((size_t)cH * cH * sizeof(bf16_t));
  bf16_t* wih1b  = (bf16_t*)take((size_t)cH * cH * sizeof(bf16_t));
  bf16_t* whh1b  = (bf16_t*)take((size_t)cH * cH * sizeof(bf16_t));
  bf16_t* woutb  = (bf16_t*)take((size_t)cO * cH * sizeof(bf16_t));
  bf16_t* h0buf  = (bf16_t*)take((size_t)2 * cHS * sizeof(bf16_t));
  bf16_t* h1buf  = (bf16_t*)take((size_t)2 * cHS * sizeof(bf16_t));
  unsigned int* bar = (unsigned int*)take(64);

  hipMemsetAsync(bar, 0, 64, stream);  // graph-capturable memset node

  // f32 -> bf16 weight/state conversion (each reused 512x afterwards).
  auto cvt = [&](const float* s, bf16_t* d, int n) {
    cvt_f32_to_bf16<<<(n + 255) / 256, 256, 0, stream>>>(s, d, n);
  };
  cvt(Whh0, whh0b, cH * cH);
  cvt(Wih1, wih1b, cH * cH);
  cvt(Whh1, whh1b, cH * cH);
  cvt(Wout, woutb, cO * cH);
  cvt(h0in,        h0buf, cHS);   // layer-0 initial state -> parity 0
  cvt(h0in + cHS,  h1buf, cHS);   // layer-1 initial state -> parity 0

  // Phase 1: big parallel WMMA GEMM for the layer-0 input projection.
  {
    const int tiles = (cBT / 16) * (cH / 16);  // 131072 tiles, 8 per block
    xp_gemm<<<tiles / 8, 256, 0, stream>>>(emb, Wih0, bih0, xp);
  }

  // Phase 2: persistent recurrent kernel, 32 resident workgroups, 226 KB LDS
  // each (gfx1250 WGP has 320 KB). Raise the dynamic-LDS cap defensively;
  // this is a host-side attribute set, not a captured stream op.
  (void)hipFuncSetAttribute((const void*)rnn_recurrent,
                            hipFuncAttributeMaxDynamicSharedMemorySize,
                            (int)LDS_BYTES);
  rnn_recurrent<<<NBLK, 256, LDS_BYTES, stream>>>(
      xp, whh0b, wih1b, whh1b, woutb, bhh0, bih1, bhh1, bout,
      h0buf, h1buf, out, bar);
}
